// MLP_60679297958438
// MI455X (gfx1250) — compile-verified, matrix operand order checked by reading, not run on hardware
//
#include <hip/hip_runtime.h>

// ---------------------------------------------------------------------------
// MI455X / gfx1250 implementation of the BiLSTM + windowed-attention network.
// bf16 WMMA (v_wmma_f32_16x16x32_bf16) for all GEMMs, f32 everywhere else.
// ---------------------------------------------------------------------------

typedef __attribute__((ext_vector_type(16))) __bf16 v16bf;
typedef __attribute__((ext_vector_type(8)))  __bf16 v8bf;
typedef __attribute__((ext_vector_type(8)))  float  v8f;

#define HID   384
#define DIN   768
#define G4H   1536
#define BATCH 16
#define SEQ   1024
#define MTOT  (BATCH * SEQ)   // 16384 rows in all batched GEMMs

// ---- WMMA fragment helpers -------------------------------------------------
// 16-bit A (16x32) per-lane layout: lanes 0-15 hold K {0..7,16..23} of row M=lane,
// lanes 16-31 hold K {8..15,24..31} of row M=lane-16.  B mirrors this with N=lane%16.
__device__ inline v16bf ld_frag(const __bf16* row, int hi) {
  const v8bf lo = *(const v8bf*)(row + (hi ? 8 : 0));
  const v8bf hv = *(const v8bf*)(row + (hi ? 24 : 16));
  v16bf r;
#pragma unroll
  for (int i = 0; i < 8; ++i) { r[i] = lo[i]; r[8 + i] = hv[i]; }
  return r;
}

__device__ inline v8f wmma_bf16(v16bf a, v16bf b, v8f c) {
  return __builtin_amdgcn_wmma_f32_16x16x32_bf16(false, a, false, b, (short)0, c,
                                                 false, false);
}

__device__ inline float sigmoidf(float x) { return 1.0f / (1.0f + __expf(-x)); }

// ---- f32 -> bf16 convert ---------------------------------------------------
__global__ void k_f32_to_bf16(const float* __restrict__ s, __bf16* __restrict__ d,
                              int n) {
  int i = blockIdx.x * blockDim.x + threadIdx.x;
  if (i < n) d[i] = (__bf16)s[i];
}

// ---- General bf16 GEMM: C = act(A[M,K] @ W[N,K]^T + bias) -------------------
// block = 256 threads (8 waves); block tile = 128(M) x 64(N); wave = 16x64 strip.
// M must be a multiple of 128 (true for all calls: M = 16384). K multiple of 32.
// N edge handling: W row index is CLAMPED (branch-free) — an out-of-range B
// column only pollutes the same out-of-range C column, which is never stored.
__global__ __launch_bounds__(256, 2)
void k_gemm_bf16(const __bf16* __restrict__ A, int lda,
                 const __bf16* __restrict__ Wt, int ldw, int N, int K,
                 const float* __restrict__ bias,
                 float* __restrict__ Cf, int ldc,
                 __bf16* __restrict__ Cbf, int ldcbf, int act) {
  const int lane = threadIdx.x & 31;
  const int wave = threadIdx.x >> 5;
  const int hi   = lane >> 4;
  const int ln   = lane & 15;
  const int m0   = blockIdx.y * 128 + wave * 16;
  const int n0   = blockIdx.x * 64;

  v8f acc[4];
#pragma unroll
  for (int t = 0; t < 4; ++t)
#pragma unroll
    for (int r = 0; r < 8; ++r) acc[t][r] = 0.0f;

  // clamped (branch-free) W row pointers, hoisted out of the k-loop
  const __bf16* wrow[4];
#pragma unroll
  for (int t = 0; t < 4; ++t) {
    int n = n0 + t * 16 + ln;
    n = n < N ? n : (N - 1);
    wrow[t] = Wt + (size_t)n * ldw;
  }
  const __bf16* arow = A + (size_t)(m0 + ln) * lda;

  for (int kb = 0; kb < K; kb += 32) {
    __builtin_prefetch(arow + kb + 64, 0, 3);   // speculative, WGP-scope
    v16bf a = ld_frag(arow + kb, hi);
#pragma unroll
    for (int t = 0; t < 4; ++t) {
      v16bf b = ld_frag(wrow[t] + kb, hi);
      acc[t] = wmma_bf16(a, b, acc[t]);
    }
  }

#pragma unroll
  for (int t = 0; t < 4; ++t) {
    const int n = n0 + t * 16 + ln;
    if (n >= N) continue;
    const float bb = bias ? bias[n] : 0.0f;
#pragma unroll
    for (int r = 0; r < 8; ++r) {
      const int m = m0 + r + 8 * hi;
      float v = acc[t][r] + bb;
      if (act == 1) v = v > 0.0f ? v : 0.0f;
      if (Cf)  Cf[(size_t)m * ldc + n] = v;
      if (Cbf) Cbf[(size_t)m * ldcbf + n] = (__bf16)v;
    }
  }
}

// ---- LSTM recurrent scan ---------------------------------------------------
// grid = 2 (blockIdx.x = direction), block = 1024 threads = 32 waves.
// Per step:  rec[16,1536] = h @ Whh^T via WMMA (96 tiles, 3/wave, K=384 in 12
// steps, accumulating from zero), gates = rec + base_gates(t).  The base-gate
// loads (coalesced, per elementwise mapping) are issued at the TOP of the step
// so they are in flight during the whole WMMA phase.  h lives in LDS (bf16,
// 12KB), recurrent pre-activations in LDS (bf16, 48KB), cell state c in
// per-thread registers.  Whh (bf16, 1.2MB) stays resident in L2.
__global__ __launch_bounds__(1024, 1)
void k_lstm_scan(const __bf16* __restrict__ gates_f,
                 const __bf16* __restrict__ gates_b,
                 const __bf16* __restrict__ Whh_f,
                 const __bf16* __restrict__ Whh_b,
                 __bf16* __restrict__ out) {
  __shared__ __bf16 sh_h[BATCH * HID];   // 12 KB
  __shared__ __bf16 sh_g[BATCH * G4H];   // 48 KB
  const int dir = blockIdx.x;
  const __bf16* gb  = dir ? gates_b : gates_f;
  const __bf16* Whh = dir ? Whh_b : Whh_f;

  const int lane = threadIdx.x & 31;
  const int wave = threadIdx.x >> 5;
  const int hi   = lane >> 4;
  const int ln   = lane & 15;
  const int n0   = (wave * 3 + 0) * 16 + ln;
  const int n1   = (wave * 3 + 1) * 16 + ln;
  const int n2   = (wave * 3 + 2) * 16 + ln;

  // per-thread elementwise mapping (fixed over time): 6 (batch, j) elements
  int eb[6], ej[6];
  float creg[6];
#pragma unroll
  for (int q = 0; q < 6; ++q) {
    const int idx = threadIdx.x + q * 1024;   // 0..6143 == 16*384
    eb[q] = idx / HID;
    ej[q] = idx % HID;
    creg[q] = 0.0f;
    sh_h[idx] = (__bf16)0.0f;
  }
  __syncthreads();

  const __bf16* wr0 = Whh + (size_t)n0 * HID;
  const __bf16* wr1 = Whh + (size_t)n1 * HID;
  const __bf16* wr2 = Whh + (size_t)n2 * HID;

  for (int t = 0; t < SEQ; ++t) {
    const int tpos = dir ? (SEQ - 1 - t) : t;

    // --- issue base-gate loads early (consumed after the barrier) ---
    float basev[6][4];
#pragma unroll
    for (int q = 0; q < 6; ++q) {
      const __bf16* brow = gb + ((size_t)eb[q] * SEQ + tpos) * G4H + ej[q];
#pragma unroll
      for (int g = 0; g < 4; ++g) basev[q][g] = (float)brow[g * HID];
    }

    // --- recurrent pre-activation: rec = h @ Whh^T (accumulate from zero) ---
    v8f a0, a1, a2;
#pragma unroll
    for (int r = 0; r < 8; ++r) { a0[r] = 0.0f; a1[r] = 0.0f; a2[r] = 0.0f; }
    for (int kb = 0; kb < HID; kb += 32) {
      v16bf a = ld_frag(&sh_h[ln * HID + kb], hi);   // A from LDS (ds path)
      a0 = wmma_bf16(a, ld_frag(wr0 + kb, hi), a0);
      a1 = wmma_bf16(a, ld_frag(wr1 + kb, hi), a1);
      a2 = wmma_bf16(a, ld_frag(wr2 + kb, hi), a2);
    }
#pragma unroll
    for (int r = 0; r < 8; ++r) {
      const int m = r + 8 * hi;
      sh_g[m * G4H + n0] = (__bf16)a0[r];
      sh_g[m * G4H + n1] = (__bf16)a1[r];
      sh_g[m * G4H + n2] = (__bf16)a2[r];
    }
    __syncthreads();

    // --- elementwise cell/hidden update (i,f,g,o gate order) ---
#pragma unroll
    for (int q = 0; q < 6; ++q) {
      const int b_ = eb[q];
      const int j  = ej[q];
      const float iv = basev[q][0] + (float)sh_g[b_ * G4H + j];
      const float fv = basev[q][1] + (float)sh_g[b_ * G4H + HID + j];
      const float gv = basev[q][2] + (float)sh_g[b_ * G4H + 2 * HID + j];
      const float ov = basev[q][3] + (float)sh_g[b_ * G4H + 3 * HID + j];
      const float c  = sigmoidf(fv) * creg[q] + sigmoidf(iv) * tanhf(gv);
      creg[q] = c;
      const float hv = sigmoidf(ov) * tanhf(c);
      sh_h[b_ * HID + j] = (__bf16)hv;
      out[((size_t)b_ * SEQ + tpos) * (2 * HID) + dir * HID + j] = (__bf16)hv;
    }
    __syncthreads();
  }
}

// ---- Banded attention with online softmax ----------------------------------
// score(i,j) = h_i.w1 + h_j.(w2 + h_i*w3) + b over |i-j|<=window.
// One wave per (b,i); lanes split the 384-d feature dim (12 each); lane
// reductions via __shfl_xor.  Output bf16 written straight into hcat[:,384:768].
__global__ __launch_bounds__(256)
void k_win_attn(const float* __restrict__ h, const float* __restrict__ attnW,
                const float* __restrict__ attnb, const int* __restrict__ win,
                __bf16* __restrict__ outbf, int ldout) {
  const int lane = threadIdx.x & 31;
  const int gw   = blockIdx.x * 8 + (threadIdx.x >> 5);
  if (gw >= MTOT) return;
  const int b  = gw / SEQ;
  const int i  = gw % SEQ;
  const int Wn = win[0];

  float w2v[12], hw3[12];
  float s1p = 0.0f;
  const float* hrow = h + (size_t)gw * HID;
#pragma unroll
  for (int q = 0; q < 12; ++q) {
    const int d  = lane + 32 * q;
    const float hv = hrow[d];
    w2v[q] = attnW[HID + d];
    hw3[q] = hv * attnW[2 * HID + d];
    s1p   += hv * attnW[d];
  }
#pragma unroll
  for (int off = 16; off > 0; off >>= 1) s1p += __shfl_xor(s1p, off, 32);
  const float bias0 = attnb[0];

  const int jlo = (i - Wn) > 0 ? (i - Wn) : 0;
  const int jhi = (i + Wn) < (SEQ - 1) ? (i + Wn) : (SEQ - 1);
  float m = -1e30f, l = 0.0f, acc[12];
#pragma unroll
  for (int q = 0; q < 12; ++q) acc[q] = 0.0f;

  for (int j = jlo; j <= jhi; ++j) {
    const float* hj = h + ((size_t)b * SEQ + j) * HID;
    float hjv[12], p = 0.0f;
#pragma unroll
    for (int q = 0; q < 12; ++q) {
      hjv[q] = hj[lane + 32 * q];
      p += hjv[q] * (w2v[q] + hw3[q]);
    }
#pragma unroll
    for (int off = 16; off > 0; off >>= 1) p += __shfl_xor(p, off, 32);
    const float sc    = s1p + p + bias0;
    const float mn    = sc > m ? sc : m;
    const float scale = __expf(m - mn);
    const float pe    = __expf(sc - mn);
    l = l * scale + pe;
#pragma unroll
    for (int q = 0; q < 12; ++q) acc[q] = acc[q] * scale + pe * hjv[q];
    m = mn;
  }
  const float inv = 1.0f / l;
  __bf16* orow = outbf + (size_t)gw * ldout;
#pragma unroll
  for (int q = 0; q < 12; ++q) orow[lane + 32 * q] = (__bf16)(acc[q] * inv);
}

// ---------------------------------------------------------------------------
extern "C" void kernel_launch(void* const* d_in, const int* in_sizes, int n_in,
                              void* d_out, int out_size, void* d_ws, size_t ws_size,
                              hipStream_t stream) {
  // setup_inputs() dict order:
  const float* x        = (const float*)d_in[0];
  const float* W1ihf    = (const float*)d_in[1];
  const float* W1hhf    = (const float*)d_in[2];
  const float* b1f      = (const float*)d_in[3];
  const float* W1ihb    = (const float*)d_in[4];
  const float* W1hhb    = (const float*)d_in[5];
  const float* b1b      = (const float*)d_in[6];
  const float* W2ihf    = (const float*)d_in[7];
  const float* W2hhf    = (const float*)d_in[8];
  const float* b2f      = (const float*)d_in[9];
  const float* W2ihb    = (const float*)d_in[10];
  const float* W2hhb    = (const float*)d_in[11];
  const float* b2b      = (const float*)d_in[12];
  const float* fc1W     = (const float*)d_in[13];
  const float* fc1b     = (const float*)d_in[14];
  const float* attnW    = (const float*)d_in[15];
  const float* attnb    = (const float*)d_in[16];
  const float* l2W1     = (const float*)d_in[17];
  const float* l2b1     = (const float*)d_in[18];
  const float* l2W2     = (const float*)d_in[19];
  const float* l2b2     = (const float*)d_in[20];
  const float* l2W3     = (const float*)d_in[21];
  const float* l2b3     = (const float*)d_in[22];
  const int*   win      = (const int*)d_in[23];

  // bump allocator over d_ws (peak ~265 MB)
  char* base = (char*)d_ws;
  size_t off = 0;
  auto alloc = [&](size_t bytes) -> void* {
    void* p = base + off;
    off = (off + bytes + 255) & ~(size_t)255;
    return p;
  };
  auto conv = [&](const float* src, size_t n) -> __bf16* {
    __bf16* d = (__bf16*)alloc(n * 2);
    k_f32_to_bf16<<<(unsigned)((n + 255) / 256), 256, 0, stream>>>(src, d, (int)n);
    return d;
  };
  auto gemm = [&](const __bf16* A, int lda, const __bf16* Wt, int ldw, int N,
                  int K, const float* bias, float* Cf, int ldc, __bf16* Cbf,
                  int ldcbf, int act) {
    dim3 grid((N + 63) / 64, MTOT / 128);
    k_gemm_bf16<<<grid, 256, 0, stream>>>(A, lda, Wt, ldw, N, K, bias, Cf, ldc,
                                          Cbf, ldcbf, act);
  };

  // bf16 conversions (weights + activations input)
  __bf16* xbf     = conv(x,     (size_t)MTOT * DIN);
  __bf16* W1ihfB  = conv(W1ihf, (size_t)G4H * DIN);
  __bf16* W1ihbB  = conv(W1ihb, (size_t)G4H * DIN);
  __bf16* W1hhfB  = conv(W1hhf, (size_t)G4H * HID);
  __bf16* W1hhbB  = conv(W1hhb, (size_t)G4H * HID);
  __bf16* W2ihfB  = conv(W2ihf, (size_t)G4H * DIN);
  __bf16* W2ihbB  = conv(W2ihb, (size_t)G4H * DIN);
  __bf16* W2hhfB  = conv(W2hhf, (size_t)G4H * HID);
  __bf16* W2hhbB  = conv(W2hhb, (size_t)G4H * HID);
  __bf16* fc1WB   = conv(fc1W,  (size_t)HID * DIN);
  __bf16* l2W1B   = conv(l2W1,  (size_t)256 * HID);
  __bf16* l2W2B   = conv(l2W2,  (size_t)96 * 256);
  __bf16* l2W3B   = conv(l2W3,  (size_t)96);

  __bf16* gates_f  = (__bf16*)alloc((size_t)MTOT * G4H * 2);   // 50 MB
  __bf16* gates_b  = (__bf16*)alloc((size_t)MTOT * G4H * 2);   // 50 MB
  __bf16* lstm1bf  = (__bf16*)alloc((size_t)MTOT * 2 * HID * 2);
  float*  h_f32    = (float*)alloc((size_t)MTOT * HID * 4);
  __bf16* hcat     = (__bf16*)alloc((size_t)MTOT * 2 * HID * 2);
  __bf16* lstm2bf  = (__bf16*)alloc((size_t)MTOT * 2 * HID * 2);
  __bf16* h2bf     = (__bf16*)alloc((size_t)MTOT * HID * 2);
  __bf16* y1bf     = (__bf16*)alloc((size_t)MTOT * 256 * 2);
  __bf16* y2bf     = (__bf16*)alloc((size_t)MTOT * 96 * 2);

  // ---- layer 1: input projections (parallel) + recurrent scan ----
  gemm(xbf, DIN, W1ihfB, DIN, G4H, DIN, b1f, nullptr, 0, gates_f, G4H, 0);
  gemm(xbf, DIN, W1ihbB, DIN, G4H, DIN, b1b, nullptr, 0, gates_b, G4H, 0);
  k_lstm_scan<<<2, 1024, 0, stream>>>(gates_f, gates_b, W1hhfB, W1hhbB, lstm1bf);

  // ---- fc1: f32 for attention + bf16 straight into hcat[:, :384] ----
  gemm(lstm1bf, 2 * HID, fc1WB, DIN, HID, DIN, fc1b, h_f32, HID, hcat, 2 * HID, 0);

  // ---- windowed attention -> bf16 into hcat[:, 384:768] ----
  k_win_attn<<<MTOT / 8, 256, 0, stream>>>(h_f32, attnW, attnb, win,
                                           hcat + HID, 2 * HID);

  // ---- layer 2 ----
  gemm(hcat, 2 * HID, W2ihfB, DIN, G4H, DIN, b2f, nullptr, 0, gates_f, G4H, 0);
  gemm(hcat, 2 * HID, W2ihbB, DIN, G4H, DIN, b2b, nullptr, 0, gates_b, G4H, 0);
  k_lstm_scan<<<2, 1024, 0, stream>>>(gates_f, gates_b, W2hhfB, W2hhbB, lstm2bf);

  // ---- head: fc1 (reused) -> MLP 384->256->96->1 ----
  gemm(lstm2bf, 2 * HID, fc1WB, DIN, HID, DIN, fc1b, nullptr, 0, h2bf, HID, 0);
  gemm(h2bf, HID, l2W1B, HID, 256, HID, l2b1, nullptr, 0, y1bf, 256, 1);
  gemm(y1bf, 256, l2W2B, 256, 96, 256, l2b2, nullptr, 0, y2bf, 96, 1);
  gemm(y2bf, 96, l2W3B, 96, 1, 96, l2b3, (float*)d_out, 1, nullptr, 0, 0);
}
